// CNN_FS_D2ST_DARM_29283087024811
// MI455X (gfx1250) — compile-verified, matrix operand order checked by reading, not run on hardware
//
#include <hip/hip_runtime.h>
#include <hip/hip_bf16.h>

typedef __attribute__((ext_vector_type(16))) _Float16 v16h;
typedef __attribute__((ext_vector_type(8)))  _Float16 v8h;
typedef __attribute__((ext_vector_type(8)))  float    v8f;

#define C_WAY 10
#define K_SHOT 5
#define NQ 400
#define T_LEN 16
#define DDIM 1024
#define W_SEG 3
#define L_SEG 8
#define NSEG 40          // K_SHOT * L_SEG
#define NSEG_PAD 48
#define QT (NQ * T_LEN)            // 6400
#define CT (C_WAY * T_LEN)         // 160
#define QV (NQ * W_SEG)            // 1200
#define CWN (C_WAY * W_SEG * NSEG_PAD) // 1440
#define CW_PAD 32                  // 30 (c,w) rows padded
#define QVL (NQ * W_SEG * L_SEG)   // 9600
#define NITEM (NQ * C_WAY * W_SEG * W_SEG) // 36000
#define ISD 0.03125f               // 1/sqrt(1024)
#define NEG_INF (-3.4e38f)

// ---------------- WMMA fragment helpers ----------------
// 16-bit A 16x32 layout: lane holds row (lane&15); K-sector base = (lane>>4)*8;
// v16h elements 0..7 = K[kb..kb+7], 8..15 = K[16+kb..16+kb+7].
__device__ inline v16h ld_frag(const _Float16* row, int lane) {
  int kb = (lane >> 4) << 3;
  v8h lo = *(const v8h*)(row + kb);
  v8h hi = *(const v8h*)(row + 16 + kb);
  return __builtin_shufflevector(lo, hi, 0,1,2,3,4,5,6,7,8,9,10,11,12,13,14,15);
}

// D[m,n] = sum_k A[m,k]*B[n,k]  (B operand fed as rows of Y = columns of B)
__device__ inline v8f gemm_rows_k1024(const _Float16* arow, const _Float16* brow, int lane) {
  v8f c = {};
#pragma unroll 8
  for (int kk = 0; kk < DDIM; kk += 32) {
    v16h a = ld_frag(arow + kk, lane);
    v16h b = ld_frag(brow + kk, lane);
    c = __builtin_amdgcn_wmma_f32_16x16x32_f16(false, a, false, b, (short)0, c, false, false);
  }
  return c;
}

// 1x2 register-blocked variant: reuse the A fragment for two N-tiles.
__device__ inline void gemm_rows_k1024_x2(const _Float16* arow, const _Float16* brow0,
                                          const _Float16* brow1, int lane,
                                          v8f& c0, v8f& c1) {
#pragma unroll 4
  for (int kk = 0; kk < DDIM; kk += 32) {
    v16h a  = ld_frag(arow  + kk, lane);
    v16h b0 = ld_frag(brow0 + kk, lane);
    v16h b1 = ld_frag(brow1 + kk, lane);
    c0 = __builtin_amdgcn_wmma_f32_16x16x32_f16(false, a, false, b0, (short)0, c0, false, false);
    c1 = __builtin_amdgcn_wmma_f32_16x16x32_f16(false, a, false, b1, (short)0, c1, false, false);
  }
}

// C/D layout: lane n = lane&15, VGPR r -> m = r + 8*(lane>>4)
__device__ inline void store_tile(float* D, int ldd, int m0, int n0, int lane, const v8f& c) {
  int n = n0 + (lane & 15);
  int mb = m0 + ((lane >> 4) << 3);
#pragma unroll
  for (int r = 0; r < 8; ++r) D[(size_t)(mb + r) * ldd + n] = c[r];
}

// ---------------- block reduction (blockDim == 256) ----------------
__device__ inline float block_sum256(float v, float* sh8) {
#pragma unroll
  for (int m = 16; m >= 1; m >>= 1) v += __shfl_xor(v, m, 32);
  int lane = threadIdx.x & 31, wid = threadIdx.x >> 5;
  if (lane == 0) sh8[wid] = v;
  __syncthreads();
  float s = 0.f;
#pragma unroll
  for (int i = 0; i < 8; ++i) s += sh8[i];
  return s;
}

__device__ inline float wave_sum(float v) {
#pragma unroll
  for (int m = 16; m >= 1; m >>= 1) v += __shfl_xor(v, m, 32);
  return v;
}
__device__ inline float wave_max(float v) {
#pragma unroll
  for (int m = 16; m >= 1; m >>= 1) v = fmaxf(v, __shfl_xor(v, m, 32));
  return v;
}

// ---------------- preprocessing (f32 exact, emit f16 operands) ----------------
// per query token: raw f16 copy + l2-normalized f16 copy
__global__ void k_prep_tokens(const float* __restrict__ q, _Float16* __restrict__ qf16,
                              _Float16* __restrict__ qnf16) {
  __shared__ float sh[8];
  size_t row = blockIdx.x;
  const float* src = q + row * DDIM;
  float v[4]; float ss = 0.f;
#pragma unroll
  for (int j = 0; j < 4; ++j) { v[j] = src[threadIdx.x + j * 256]; ss += v[j] * v[j]; }
  ss = block_sum256(ss, sh);
  float rinv = 1.f / fmaxf(sqrtf(ss), 1e-12f);
#pragma unroll
  for (int j = 0; j < 4; ++j) {
    int i = threadIdx.x + j * 256;
    qf16[row * DDIM + i]  = (_Float16)v[j];
    qnf16[row * DDIM + i] = (_Float16)(v[j] * rinv);
  }
}

// prototype = mean over shots, then l2norm -> pn f16 [160,1024]
__global__ void k_prep_proto(const float* __restrict__ sup, _Float16* __restrict__ pn) {
  __shared__ float sh[8];
  int row = blockIdx.x, c = row / 16, t = row % 16;
  float v[4] = {0,0,0,0};
#pragma unroll
  for (int k = 0; k < K_SHOT; ++k) {
    const float* r = sup + ((size_t)(c * K_SHOT + k) * 16 + t) * DDIM;
#pragma unroll
    for (int j = 0; j < 4; ++j) v[j] += r[threadIdx.x + j * 256];
  }
  float ss = 0.f;
#pragma unroll
  for (int j = 0; j < 4; ++j) { v[j] *= 0.2f; ss += v[j] * v[j]; }
  ss = block_sum256(ss, sh);
  float rinv = 1.f / fmaxf(sqrtf(ss), 1e-12f);
#pragma unroll
  for (int j = 0; j < 4; ++j)
    pn[(size_t)row * DDIM + threadIdx.x + j * 256] = (_Float16)(v[j] * rinv);
}

// q_cond = l2norm(mean over window tokens) -> f16 [1200,1024]
__global__ void k_prep_qcond(const float* __restrict__ q, _Float16* __restrict__ qc) {
  __shared__ float sh[8];
  int g = blockIdx.x, qi = g / 3, w = g % 3, st = 4 * w;
  const float* base = q + ((size_t)qi * 16 + st) * DDIM;
  float v[4] = {0,0,0,0};
#pragma unroll
  for (int l = 0; l < L_SEG; ++l)
#pragma unroll
    for (int j = 0; j < 4; ++j) v[j] += base[(size_t)l * DDIM + threadIdx.x + j * 256];
  float ss = 0.f;
#pragma unroll
  for (int j = 0; j < 4; ++j) { v[j] *= 0.125f; ss += v[j] * v[j]; }
  ss = block_sum256(ss, sh);
  float rinv = 1.f / fmaxf(sqrtf(ss), 1e-12f);
#pragma unroll
  for (int j = 0; j < 4; ++j)
    qc[(size_t)g * DDIM + threadIdx.x + j * 256] = (_Float16)(v[j] * rinv);
}

// cls_sup tokens gathered to f16 [30][48][1024], rows >=40 zero-padded
__global__ void k_prep_clssup(const float* __restrict__ sup, _Float16* __restrict__ cs) {
  int b = blockIdx.x, cw = b / NSEG_PAD, n = b % NSEG_PAD;
  _Float16* dst = cs + (size_t)b * DDIM;
  if (n >= NSEG) {
#pragma unroll
    for (int j = 0; j < 4; ++j) dst[threadIdx.x + j * 256] = (_Float16)0.f;
    return;
  }
  int c = cw / 3, w = cw % 3, k = n / L_SEG, l = n % L_SEG;
  const float* src = sup + ((size_t)(c * K_SHOT + k) * 16 + 4 * w + l) * DDIM;
#pragma unroll
  for (int j = 0; j < 4; ++j)
    dst[threadIdx.x + j * 256] = (_Float16)src[threadIdx.x + j * 256];
}

// sup_cond = l2norm(mean over 40 tokens) -> f16 [32,1024], rows 30/31 zero
__global__ void k_prep_supcond(const float* __restrict__ sup, _Float16* __restrict__ sc) {
  __shared__ float sh[8];
  int b = blockIdx.x;
  if (b >= 30) {
#pragma unroll
    for (int j = 0; j < 4; ++j) sc[(size_t)b * DDIM + threadIdx.x + j * 256] = (_Float16)0.f;
    return;
  }
  int c = b / 3, w = b % 3, st = 4 * w;
  float v[4] = {0,0,0,0};
  for (int k = 0; k < K_SHOT; ++k)
    for (int l = 0; l < L_SEG; ++l) {
      const float* r = sup + ((size_t)(c * K_SHOT + k) * 16 + st + l) * DDIM;
#pragma unroll
      for (int j = 0; j < 4; ++j) v[j] += r[threadIdx.x + j * 256];
    }
  float ss = 0.f;
#pragma unroll
  for (int j = 0; j < 4; ++j) { v[j] *= (1.f / 40.f); ss += v[j] * v[j]; }
  ss = block_sum256(ss, sh);
  float rinv = 1.f / fmaxf(sqrtf(ss), 1e-12f);
#pragma unroll
  for (int j = 0; j < 4; ++j)
    sc[(size_t)b * DDIM + threadIdx.x + j * 256] = (_Float16)(v[j] * rinv);
}

// ---------------- WMMA GEMMs (blockDim 128 = 4 waves) ----------------
// sim: [6400,1024] x [160,1024]^T, 1x2 N-blocked: 400 M-tiles x 5 N-pairs
__global__ void k_gemm_sim(const _Float16* __restrict__ qn, const _Float16* __restrict__ pn,
                           float* __restrict__ sim) {
  int lane = threadIdx.x & 31, wid = threadIdx.x >> 5;
  int tile = blockIdx.x * 4 + wid;                // 2000 tiles, exact
  int mt = tile / 5, np = tile % 5;
  const _Float16* ar  = qn + (size_t)(mt * 16 + (lane & 15)) * DDIM;
  const _Float16* br0 = pn + (size_t)(np * 32 + (lane & 15)) * DDIM;
  const _Float16* br1 = br0 + (size_t)16 * DDIM;
  v8f c0 = {}, c1 = {};
  gemm_rows_k1024_x2(ar, br0, br1, lane, c0, c1);
  store_tile(sim, CT, mt * 16, np * 32,      lane, c0);
  store_tile(sim, CT, mt * 16, np * 32 + 16, lane, c1);
}

// S raw scores: [1200,1024] x [1440,1024]^T, 1x2 N-blocked: 75 x 45 pairs
__global__ void k_gemm_S(const _Float16* __restrict__ qc, const _Float16* __restrict__ cs,
                         float* __restrict__ S) {
  int lane = threadIdx.x & 31, wid = threadIdx.x >> 5;
  int tile = blockIdx.x * 4 + wid;
  if (tile >= (QV / 16) * (CWN / 32)) return;     // 75*45 = 3375
  int mt = tile / (CWN / 32), np = tile % (CWN / 32);
  const _Float16* ar  = qc + (size_t)(mt * 16 + (lane & 15)) * DDIM;
  const _Float16* br0 = cs + (size_t)(np * 32 + (lane & 15)) * DDIM;
  const _Float16* br1 = br0 + (size_t)16 * DDIM;
  v8f c0 = {}, c1 = {};
  gemm_rows_k1024_x2(ar, br0, br1, lane, c0, c1);
  store_tile(S, CWN, mt * 16, np * 32,      lane, c0);
  store_tile(S, CWN, mt * 16, np * 32 + 16, lane, c1);
}

// Qraw: [9600,1024] x [32,1024]^T, both N-tiles per wave: 600 tiles
__global__ void k_gemm_qraw(const _Float16* __restrict__ qf, const _Float16* __restrict__ sc,
                            float* __restrict__ Qr) {
  int lane = threadIdx.x & 31, wid = threadIdx.x >> 5;
  int mt = blockIdx.x * 4 + wid;                  // 600 tiles, exact
  int mf = mt * 16 + (lane & 15);                 // flat (q,v,n)
  int qi = mf / 24, rr = mf % 24, v = rr / L_SEG, n = rr % L_SEG;
  const _Float16* ar  = qf + ((size_t)qi * 16 + 4 * v + n) * DDIM;
  const _Float16* br0 = sc + (size_t)(lane & 15) * DDIM;
  const _Float16* br1 = br0 + (size_t)16 * DDIM;
  v8f c0 = {}, c1 = {};
  gemm_rows_k1024_x2(ar, br0, br1, lane, c0, c1);
  store_tile(Qr, CW_PAD, mt * 16, 0,  lane, c0);
  store_tile(Qr, CW_PAD, mt * 16, 16, lane, c1);
}

__global__ void k_gemm_gsup(const _Float16* __restrict__ cs, float* __restrict__ G) {
  int lane = threadIdx.x & 31, wid = threadIdx.x >> 5;
  int tile = blockIdx.x * 4 + wid;
  if (tile >= 30 * 9) return;
  int cw = tile / 9, sub = tile % 9, mt = sub / 3, nt = sub % 3;
  const _Float16* ar = cs + (size_t)(cw * NSEG_PAD + mt * 16 + (lane & 15)) * DDIM;
  const _Float16* br = cs + (size_t)(cw * NSEG_PAD + nt * 16 + (lane & 15)) * DDIM;
  v8f c = gemm_rows_k1024(ar, br, lane);
  store_tile(G + (size_t)cw * NSEG_PAD * NSEG_PAD, NSEG_PAD, mt * 16, nt * 16, lane, c);
}

// two (q,v) 8x8 Grams packed per 16x16 tile; keep diagonal blocks only
__global__ void k_gemm_gq(const _Float16* __restrict__ qf, float* __restrict__ Gq) {
  int lane = threadIdx.x & 31, wid = threadIdx.x >> 5;
  int p = blockIdx.x * 4 + wid;                   // 600 tiles, exact
  int g0 = 2 * p;
  int local = lane & 15;
  int g = g0 + (local >> 3), n = local & 7;
  int qi = g / 3, v = g % 3;
  const _Float16* row = qf + ((size_t)qi * 16 + 4 * v + n) * DDIM;
  v8f c = gemm_rows_k1024(row, row, lane);        // X * X^T
  int ncol = lane & 15, mb = (lane >> 4) << 3;
#pragma unroll
  for (int r = 0; r < 8; ++r) {
    int m = mb + r;
    if ((m >> 3) == (ncol >> 3))                  // diagonal 8x8 block
      Gq[(size_t)(g0 + (m >> 3)) * 64 + (m & 7) * 8 + (ncol & 7)] = c[r];
  }
}

// ---------------- reductions ----------------
// bi-MHM: score = sum_t max_s sim + sum_s max_t sim - 2T
__global__ void k_global(const float* __restrict__ sim, float* __restrict__ gsc) {
  int id = blockIdx.x * 256 + threadIdx.x;
  if (id >= NQ * C_WAY) return;
  int q = id / C_WAY, c = id % C_WAY;
  const float* S = sim + (size_t)q * 16 * CT + c * 16;
  float acc = -32.f;
  for (int t = 0; t < 16; ++t) {
    float mx = NEG_INF;
    for (int s = 0; s < 16; ++s) mx = fmaxf(mx, S[(size_t)t * CT + s]);
    acc += mx;
  }
  for (int s = 0; s < 16; ++s) {
    float mx = NEG_INF;
    for (int t = 0; t < 16; ++t) mx = fmaxf(mx, S[(size_t)t * CT + s]);
    acc += mx;
  }
  gsc[id] = acc;
}

// s2q: one wave per (q,c,v,w); softmax over 40, num = sum a*raw, den^2 = a^T G a
__global__ void k_s2q(const float* __restrict__ S, const float* __restrict__ G,
                      float* __restrict__ out) {
  __shared__ float sh[4][48];
  int lane = threadIdx.x & 31, wid = threadIdx.x >> 5;
  int item = blockIdx.x * 4 + wid;                // 36000, exact
  int q = item / 90, r = item % 90, c = r / 9, s = r % 9, v = s / 3, w = s % 3;
  size_t base = (size_t)(q * 3 + v) * CWN + (size_t)(c * 3 + w) * NSEG_PAD;
  bool v1 = (lane < 8);
  float r0 = S[base + lane];
  float r1 = v1 ? S[base + lane + 32] : 0.f;
  float s0 = r0 * ISD, s1 = v1 ? r1 * ISD : NEG_INF;
  float mx = wave_max(fmaxf(s0, s1));
  float e0 = expf(s0 - mx), e1 = v1 ? expf(s1 - mx) : 0.f;
  float inv = 1.f / wave_sum(e0 + e1);
  float a0 = e0 * inv, a1 = e1 * inv;
  float num = wave_sum(a0 * r0 + a1 * r1);
  sh[wid][lane] = a0;
  if (v1) sh[wid][lane + 32] = a1;
  __syncthreads();
  const float* Gr = G + (size_t)(c * 3 + w) * NSEG_PAD * NSEG_PAD;
  float t0 = 0.f, t1 = 0.f;
  for (int m = 0; m < NSEG; ++m) t0 += sh[wid][m] * Gr[(size_t)lane * NSEG_PAD + m];
  if (v1) for (int m = 0; m < NSEG; ++m) t1 += sh[wid][m] * Gr[(size_t)(lane + 32) * NSEG_PAD + m];
  float den2 = wave_sum(a0 * t0 + a1 * t1);
  if (lane == 0) out[item] = num / fmaxf(sqrtf(den2), 1e-12f);
}

// q2s: one thread per (q,c,v,w); softmax over 8, Gram 8x8 quadratic form
__global__ void k_q2s(const float* __restrict__ Qr, const float* __restrict__ Gq,
                      float* __restrict__ out) {
  int id = blockIdx.x * 256 + threadIdx.x;
  if (id >= NITEM) return;
  int q = id / 90, r = id % 90, c = r / 9, s = r % 9, v = s / 3, w = s % 3;
  int g = q * 3 + v, col = c * 3 + w;
  float raw[8], a[8];
  float mx = NEG_INF;
#pragma unroll
  for (int n = 0; n < 8; ++n) {
    raw[n] = Qr[((size_t)g * 8 + n) * CW_PAD + col];
    mx = fmaxf(mx, raw[n] * ISD);
  }
  float tot = 0.f;
#pragma unroll
  for (int n = 0; n < 8; ++n) { a[n] = expf(raw[n] * ISD - mx); tot += a[n]; }
  float inv = 1.f / tot, num = 0.f;
#pragma unroll
  for (int n = 0; n < 8; ++n) { a[n] *= inv; num += a[n] * raw[n]; }
  const float* G = Gq + (size_t)g * 64;
  float den2 = 0.f;
#pragma unroll
  for (int n = 0; n < 8; ++n) {
    float t = 0.f;
#pragma unroll
    for (int m = 0; m < 8; ++m) t += a[m] * G[n * 8 + m];
    den2 += a[n] * t;
  }
  out[id] = num / fmaxf(sqrtf(den2), 1e-12f);
}

// top-2 mean of 9, fuse branches
__global__ void k_final(const float* __restrict__ gsc, const float* __restrict__ s2q,
                        const float* __restrict__ q2s, const float* __restrict__ flog,
                        float* __restrict__ out) {
  int id = blockIdx.x * 256 + threadIdx.x;
  if (id >= NQ * C_WAY) return;
  float l0 = flog[0], l1 = flog[1];
  float m = fmaxf(l0, l1);
  float e0 = expf(l0 - m), e1 = expf(l1 - m), inv = 1.f / (e0 + e1);
  float f0 = e0 * inv, f1 = e1 * inv;
  const float* a = s2q + (size_t)id * 9;
  float b1 = NEG_INF, b2 = NEG_INF;
  for (int i = 0; i < 9; ++i) { float x = a[i]; if (x > b1) { b2 = b1; b1 = x; } else if (x > b2) b2 = x; }
  const float* b = q2s + (size_t)id * 9;
  float c1 = NEG_INF, c2 = NEG_INF;
  for (int i = 0; i < 9; ++i) { float x = b[i]; if (x > c1) { c2 = c1; c1 = x; } else if (x > c2) c2 = x; }
  float seg = 0.5f * (0.5f * (b1 + b2) + 0.5f * (c1 + c2));
  out[id] = f0 * gsc[id] + f1 * seg;
}

extern "C" void kernel_launch(void* const* d_in, const int* in_sizes, int n_in,
                              void* d_out, int out_size, void* d_ws, size_t ws_size,
                              hipStream_t stream) {
  (void)in_sizes; (void)n_in; (void)out_size; (void)ws_size;
  const float* sup  = (const float*)d_in[0];
  const float* qry  = (const float*)d_in[1];
  const float* flog = (const float*)d_in[2];
  float* out = (float*)d_out;

  char* w = (char*)d_ws;
  size_t off = 0;
  auto carve = [&](size_t bytes) -> void* {
    void* p = w + off;
    off += (bytes + 255) & ~(size_t)255;
    return p;
  };

  _Float16* q_f16  = (_Float16*)carve((size_t)QT * DDIM * 2);
  _Float16* qn_f16 = (_Float16*)carve((size_t)QT * DDIM * 2);
  _Float16* pn_f16 = (_Float16*)carve((size_t)CT * DDIM * 2);
  _Float16* qc_f16 = (_Float16*)carve((size_t)QV * DDIM * 2);
  _Float16* cs_f16 = (_Float16*)carve((size_t)CWN * DDIM * 2);
  _Float16* sc_f16 = (_Float16*)carve((size_t)CW_PAD * DDIM * 2);
  float* sim  = (float*)carve((size_t)QT * CT * 4);
  float* Sraw = (float*)carve((size_t)QV * CWN * 4);
  float* Qraw = (float*)carve((size_t)QVL * CW_PAD * 4);
  float* Gsup = (float*)carve((size_t)30 * NSEG_PAD * NSEG_PAD * 4);
  float* Gq   = (float*)carve((size_t)QV * 64 * 4);
  float* gsc  = (float*)carve((size_t)NQ * C_WAY * 4);
  float* s2q  = (float*)carve((size_t)NITEM * 4);
  float* q2s  = (float*)carve((size_t)NITEM * 4);

  // preprocessing (f32 exact math, f16 operand emission)
  k_prep_tokens <<<QT,     256, 0, stream>>>(qry, q_f16, qn_f16);
  k_prep_proto  <<<CT,     256, 0, stream>>>(sup, pn_f16);
  k_prep_qcond  <<<QV,     256, 0, stream>>>(qry, qc_f16);
  k_prep_clssup <<<CWN,    256, 0, stream>>>(sup, cs_f16);
  k_prep_supcond<<<CW_PAD, 256, 0, stream>>>(sup, sc_f16);

  // WMMA GEMMs (4 waves / block; large GEMMs 1x2 register-blocked in N)
  k_gemm_sim <<<(QT/16)*(CT/32)/4,            128, 0, stream>>>(qn_f16, pn_f16, sim);   // 500
  k_gemm_S   <<<((QV/16)*(CWN/32) + 3) / 4,   128, 0, stream>>>(qc_f16, cs_f16, Sraw);  // 844
  k_gemm_qraw<<<(QVL/16)/4,                   128, 0, stream>>>(q_f16, sc_f16, Qraw);   // 150
  k_gemm_gsup<<<(30*9 + 3) / 4,               128, 0, stream>>>(cs_f16, Gsup);          // 68
  k_gemm_gq  <<<(QV/2)/4,                     128, 0, stream>>>(q_f16, Gq);             // 150

  // score reductions
  k_global<<<(NQ*C_WAY + 255) / 256, 256, 0, stream>>>(sim, gsc);
  k_s2q   <<<NITEM / 4,              128, 0, stream>>>(Sraw, Gsup, s2q);
  k_q2s   <<<(NITEM + 255) / 256,    256, 0, stream>>>(Qraw, Gq, q2s);
  k_final <<<(NQ*C_WAY + 255) / 256, 256, 0, stream>>>(gsc, s2q, q2s, flog, out);
}
